// GNN_74483322847536
// MI455X (gfx1250) — matrix operand order repaired and checked by reading, not compiled, not run on hardware
//
#include <hip/hip_runtime.h>

#define N_NODES 100000
#define N_EDGES 640000
#define DIM     128
#define NLAYERS 5
#define BN_EPS  1e-5f

typedef _Float16 half_t;
typedef __attribute__((ext_vector_type(16))) _Float16 v16h;
typedef __attribute__((ext_vector_type(8)))  _Float16 v8h;
typedef __attribute__((ext_vector_type(8)))  float    v8f;

// ---------------- elementwise helpers ----------------

__global__ void zero_f32(float* __restrict__ p, int n) {
    int i = blockIdx.x * blockDim.x + threadIdx.x;
    if (i < n) p[i] = 0.0f;
}

__global__ void cvt_f32_to_f16(const float* __restrict__ in, half_t* __restrict__ out, int n) {
    int i = blockIdx.x * blockDim.x + threadIdx.x;
    if (i < n) out[i] = (half_t)in[i];
}

// agg[dst[e]*D + f] += h[src[e]*D + f]; one thread handles 4 features of one edge
__global__ void scatter_add(const float* __restrict__ h,
                            const long long* __restrict__ src,
                            const long long* __restrict__ dst,
                            float* __restrict__ agg) {
    int tid = blockIdx.x * blockDim.x + threadIdx.x;     // E * 32 threads
    int e = tid >> 5;
    if (e >= N_EDGES) return;
    int f = (tid & 31) * 4;
    int s = (int)src[e];
    int d = (int)dst[e];
    const float4 v = *(const float4*)(h + (size_t)s * DIM + f);
    float* a = agg + (size_t)d * DIM + f;
    atomicAdd(a + 0, v.x);
    atomicAdd(a + 1, v.y);
    atomicAdd(a + 2, v.z);
    atomicAdd(a + 3, v.w);
}

// t16 = (f16)(h + agg)
__global__ void add_cvt(const float* __restrict__ h, const float* __restrict__ agg,
                        half_t* __restrict__ t16, int n) {
    int i = blockIdx.x * blockDim.x + threadIdx.x;
    if (i < n) t16[i] = (half_t)(h[i] + agg[i]);
}

// column sums + sums of squares over the node dim: stats[0..127]=sum, stats[128..255]=sumsq
__global__ void col_stats(const float* __restrict__ y, float* __restrict__ stats) {
    int j  = threadIdx.x;            // 128 threads = one per column
    int r0 = blockIdx.x * 250;       // 400 blocks * 250 rows = 100000
    float s = 0.0f, q = 0.0f;
    for (int r = 0; r < 250; ++r) {
        float v = y[(size_t)(r0 + r) * DIM + j];
        s += v;
        q += v * v;
    }
    atomicAdd(&stats[j], s);
    atomicAdd(&stats[DIM + j], q);
}

// f16out = relu(BN(y)) -- feeds the next GEMM
__global__ void bn_relu_cvt(const float* __restrict__ y, const float* __restrict__ stats,
                            const float* __restrict__ gamma, const float* __restrict__ beta,
                            half_t* __restrict__ out, int n) {
    int i = blockIdx.x * blockDim.x + threadIdx.x;
    if (i >= n) return;
    int j = i & (DIM - 1);
    const float inv_n = 1.0f / (float)N_NODES;
    float mu  = stats[j] * inv_n;
    float var = stats[DIM + j] * inv_n - mu * mu;
    float sc  = gamma[j] * rsqrtf(var + BN_EPS);
    float v   = (y[i] - mu) * sc + beta[j];
    out[i] = (half_t)fmaxf(v, 0.0f);
}

// hout = relu(BN(y)) + x  (residual to original input), f32
__global__ void bn_relu_add(const float* __restrict__ y, const float* __restrict__ stats,
                            const float* __restrict__ gamma, const float* __restrict__ beta,
                            const float* __restrict__ x, float* __restrict__ hout, int n) {
    int i = blockIdx.x * blockDim.x + threadIdx.x;
    if (i >= n) return;
    int j = i & (DIM - 1);
    const float inv_n = 1.0f / (float)N_NODES;
    float mu  = stats[j] * inv_n;
    float var = stats[DIM + j] * inv_n - mu * mu;
    float sc  = gamma[j] * rsqrtf(var + BN_EPS);
    float v   = (y[i] - mu) * sc + beta[j];
    hout[i] = fmaxf(v, 0.0f) + x[i];
}

// ---------------- WMMA GEMM: Y[N x 128] = X16[N x 128] @ W16^T ----------------
// W16 is row-major [j][k] (original torch Linear weight layout), which is exactly
// the B column (K x N) each lane needs in this code's contiguous k order (same order used for A).
__global__ __launch_bounds__(256) void gemm_wmma(const half_t* __restrict__ X,
                                                 const half_t* __restrict__ W,
                                                 float* __restrict__ Y) {
    const int lane = threadIdx.x & 31;
    const int wave = threadIdx.x >> 5;       // 8 waves -> 8 column tiles
    const int row0 = blockIdx.x * 16;        // 6250 row tiles
    const int col0 = wave * 16;
    const int m = lane & 15;                 // A row within tile / D column
    const int g = lane >> 4;                 // lane half-group

    const half_t* arow = X + (size_t)(row0 + m) * DIM;
    const half_t* brow = W + (size_t)(col0 + m) * DIM + 16 * g;  // j = col0 + (lane&15)

    v8f acc = {};

#pragma unroll
    for (int kk = 0; kk < 4; ++kk) {         // K = 128 in steps of 32
        // A: halves 0..15 -> K = kk*32 + 16g .. +16 (contiguous, matches B)
        v8h alo = *(const v8h*)(arow + kk * 32 + 16 * g);
        v8h ahi = *(const v8h*)(arow + kk * 32 + 16 * g + 8);
        v16h a;
#pragma unroll
        for (int i = 0; i < 8; ++i) { a[i] = alo[i]; a[i + 8] = ahi[i]; }
        // B: halves 0..15 -> K = kk*32 + 16g .. +16 (contiguous in k of row-major W[j][k])
        v16h b = *(const v16h*)(brow + kk * 32);
        acc = __builtin_amdgcn_wmma_f32_16x16x32_f16(
            /*neg_a=*/false, a, /*neg_b=*/false, b,
            /*c_mod=*/(short)0, acc, /*reuse_a=*/false, /*reuse_b=*/false);
    }

    // D: VGPR r holds row (row0 + r + 8g), column col0 + (lane&15)
    float* yout = Y + (size_t)(row0 + 8 * g) * DIM + col0 + m;
#pragma unroll
    for (int r = 0; r < 8; ++r) yout[(size_t)r * DIM] = acc[r];
}

// ---------------- driver ----------------

extern "C" void kernel_launch(void* const* d_in, const int* in_sizes, int n_in,
                              void* d_out, int out_size, void* d_ws, size_t ws_size,
                              hipStream_t stream) {
    const float*     x    = (const float*)d_in[0];
    const long long* eidx = (const long long*)d_in[1];   // int64 (2, E)
    const float*     W1   = (const float*)d_in[2];       // (L, D, D)
    // d_in[3] = b1 (cancels inside BN), d_in[5]... per reference order
    const float*     g1   = (const float*)d_in[4];
    const float*     be1  = (const float*)d_in[5];
    const float*     W2   = (const float*)d_in[6];
    const float*     g2   = (const float*)d_in[8];
    const float*     be2  = (const float*)d_in[9];
    float*           out  = (float*)d_out;

    const long long* src = eidx;
    const long long* dst = eidx + N_EDGES;

    const int ND  = N_NODES * DIM;           // 12,800,000
    const int DD  = DIM * DIM;               // 16,384
    const int LDD = NLAYERS * DD;

    // workspace layout (256B-aligned offsets)
    char* ws = (char*)d_ws;
    float*  h_ws  = (float*)(ws + 0);                                 // 51.2 MB
    float*  y     = (float*)(ws + (size_t)ND * 4);                    // 51.2 MB (agg & gemm out)
    half_t* t16   = (half_t*)(ws + (size_t)ND * 8);                   // 25.6 MB
    half_t* w1_16 = (half_t*)(ws + (size_t)ND * 10);                  // 160 KB
    half_t* w2_16 = (half_t*)(ws + (size_t)ND * 10 + (size_t)LDD * 2);
    float*  stats = (float*)(ws + (size_t)ND * 10 + (size_t)LDD * 4); // 1 KB

    const int TB = 256;
    const int ew_blocks  = (ND + TB - 1) / TB;               // 50000
    const int sc_blocks  = (N_EDGES * 32 + TB - 1) / TB;     // 80000
    const int cvt_blocks = (LDD + TB - 1) / TB;

    // one-time (per call) weight conversion to f16
    cvt_f32_to_f16<<<cvt_blocks, TB, 0, stream>>>(W1, w1_16, LDD);
    cvt_f32_to_f16<<<cvt_blocks, TB, 0, stream>>>(W2, w2_16, LDD);

    for (int l = 0; l < NLAYERS; ++l) {
        const float* h_in  = (l == 0) ? x : h_ws;
        float*       h_out = (l == NLAYERS - 1) ? out : h_ws;

        // agg = segment_sum(h[src], dst)
        zero_f32<<<ew_blocks, TB, 0, stream>>>(y, ND);       // y doubles as agg
        scatter_add<<<sc_blocks, TB, 0, stream>>>(h_in, src, dst, y);

        // t = h + agg  (f16 for WMMA)
        add_cvt<<<ew_blocks, TB, 0, stream>>>(h_in, y, t16, ND);

        // Linear1 -> BN -> relu  (bias cancels in BN)
        gemm_wmma<<<N_NODES / 16, TB, 0, stream>>>(t16, w1_16 + (size_t)l * DD, y);
        zero_f32<<<1, 256, 0, stream>>>(stats, 2 * DIM);
        col_stats<<<N_NODES / 250, DIM, 0, stream>>>(y, stats);
        bn_relu_cvt<<<ew_blocks, TB, 0, stream>>>(y, stats, g1 + (size_t)l * DIM,
                                                  be1 + (size_t)l * DIM, t16, ND);

        // Linear2 -> BN -> relu -> + x
        gemm_wmma<<<N_NODES / 16, TB, 0, stream>>>(t16, w2_16 + (size_t)l * DD, y);
        zero_f32<<<1, 256, 0, stream>>>(stats, 2 * DIM);
        col_stats<<<N_NODES / 250, DIM, 0, stream>>>(y, stats);
        bn_relu_add<<<ew_blocks, TB, 0, stream>>>(y, stats, g2 + (size_t)l * DIM,
                                                  be2 + (size_t)l * DIM, x, h_out, ND);
    }
}
